// SinkhornTransformerBlock_51144470560899
// MI455X (gfx1250) — compile-verified, matrix operand order checked by reading, not run on hardware
//
#include <hip/hip_runtime.h>
#include <hip/hip_bf16.h>

// ---------- constants ----------
#define B_ 4
#define S_ 4096
#define D_ 1024
#define H_ 16
#define DH_ 64
#define BLK_ 128
#define NB_ 32
#define MTOK (B_ * S_)          // 16384
#define LN_EPS 1e-6f

// ---------- WMMA vector types ----------
typedef __attribute__((ext_vector_type(16))) __bf16 v16bf;
typedef __attribute__((ext_vector_type(8)))  __bf16 v8bf;
typedef __attribute__((ext_vector_type(8)))  float  v8f;
typedef __attribute__((ext_vector_type(4)))  unsigned u32x4;
typedef __attribute__((ext_vector_type(8)))  unsigned u32x8;

__device__ inline v16bf make_frag(v8bf lo, v8bf hi) {
  return __builtin_shufflevector(lo, hi, 0,1,2,3,4,5,6,7,8,9,10,11,12,13,14,15);
}

// A/B fragment loader for 16-bit WMMA operands stored row-major in LDS.
// Lane half (lane>=16) selects K groups {8..15, 24..31}; else {0..7, 16..23}.
__device__ inline v16bf frag_ld_bf(const __bf16* base, int stride, int row, int kbase, int half) {
  v8bf lo = *(const v8bf*)(base + row * stride + kbase + half * 8);
  v8bf hi = *(const v8bf*)(base + row * stride + kbase + 16 + half * 8);
  return make_frag(lo, hi);
}

// Same, but source is fp32 in LDS (softmax probabilities) converted to bf16.
__device__ inline v16bf frag_ld_f32(const float* base, int stride, int row, int kbase, int half) {
  const float* p0 = base + row * stride + kbase + half * 8;
  const float* p1 = base + row * stride + kbase + 16 + half * 8;
  v16bf r;
#pragma unroll
  for (int i = 0; i < 8; ++i) {
    r[i]     = (__bf16)p0[i];
    r[i + 8] = (__bf16)p1[i];
  }
  return r;
}

// ---------- CDNA5 async copy: global -> LDS, 16 bytes per lane ----------
// GV addressing: VDST = per-lane LDS byte offset, VADDR = 64-bit global address.
__device__ inline void async_copy_b128(unsigned lds_byte_off, const void* gptr) {
  asm volatile("global_load_async_to_lds_b128 %0, %1, off"
               :: "v"(lds_byte_off), "v"(gptr)
               : "memory");
}
__device__ inline void wait_asynccnt0() {
  asm volatile("s_wait_asynccnt 0x0" ::: "memory");
}

// ---------- CDNA5 Tensor Data Mover: 2D tile DMA global -> LDS ----------
// Builds D# group0 (128b) + group1 (256b); groups 2/3 zero (<=2D tensor).
// tile: tile_h rows x tile_w elements, row stride = stride_elems, 2-byte data.
__device__ inline void tdm_load_2d_bf16(unsigned lds_byte, const void* gptr,
                                        unsigned tile_w, unsigned tile_h,
                                        unsigned stride_elems) {
  unsigned long long ga = (unsigned long long)gptr;
  u32x4 g0;
  g0[0] = 1u;                                   // count=1, user descriptor
  g0[1] = lds_byte;                             // lds_addr
  g0[2] = (unsigned)ga;                         // global_addr[31:0]
  g0[3] = (unsigned)(ga >> 32) | (2u << 30);    // global_addr[56:32] | type=2
  const unsigned dim0 = 0x7fffffffu, dim1 = 0x7fffffffu; // huge: no OOB clip
  u32x8 g1;
  g1[0] = 0x00010000u;                          // data_size=1 (2 bytes)
  g1[1] = (dim0 & 0xffffu) << 16;               // tensor_dim0 lo
  g1[2] = (dim0 >> 16) | ((dim1 & 0xffffu) << 16);
  g1[3] = (dim1 >> 16) | (tile_w << 16);        // tile_dim0
  g1[4] = tile_h;                               // tile_dim1 (tile_dim2=0)
  g1[5] = stride_elems;                         // tensor_dim0_stride lo
  g1[6] = 0u;                                   // stride hi | dim1_stride lo
  g1[7] = 0u;
  u32x4 gz = {0u, 0u, 0u, 0u};
  asm volatile("tensor_load_to_lds %0, %1, %2, %3"
               :: "s"(g0), "s"(g1), "s"(gz), "s"(gz)
               : "memory");
}

__device__ inline float gelu_tanh(float x) {
  const float k0 = 0.7978845608028654f; // sqrt(2/pi)
  const float k1 = 0.044715f;
  float x3 = x * x * x;
  return 0.5f * x * (1.0f + tanhf(k0 * (x + k1 * x3)));
}

// ---------- elementwise fp32 -> bf16 ----------
__global__ void cvt_f32_bf16(const float* __restrict__ src, __bf16* __restrict__ dst, int n) {
  int i = blockIdx.x * blockDim.x + threadIdx.x;
  int stride = gridDim.x * blockDim.x;
  for (; i < n; i += stride) dst[i] = (__bf16)src[i];
}

// ---------- LayerNorm over D=1024, output bf16 ----------
__global__ __launch_bounds__(256) void layernorm_bf16(
    const float* __restrict__ x, const float* __restrict__ scale,
    const float* __restrict__ bias, __bf16* __restrict__ out) {
  const int row = blockIdx.x;
  const int tid = threadIdx.x;
  const float* xr = x + (size_t)row * D_;
  float s = 0.f, s2 = 0.f;
#pragma unroll
  for (int i = 0; i < 4; ++i) {
    float v = xr[tid + i * 256];
    s += v; s2 += v * v;
  }
  __shared__ float red[256], red2[256];
  red[tid] = s; red2[tid] = s2;
  __syncthreads();
  for (int off = 128; off > 0; off >>= 1) {
    if (tid < off) { red[tid] += red[tid + off]; red2[tid] += red2[tid + off]; }
    __syncthreads();
  }
  float mu  = red[0] * (1.0f / D_);
  float var = red2[0] * (1.0f / D_) - mu * mu;
  float inv = rsqrtf(var + LN_EPS);
#pragma unroll
  for (int i = 0; i < 4; ++i) {
    int c = tid + i * 256;
    float v = (xr[c] - mu) * inv * scale[c] + bias[c];
    out[(size_t)row * D_ + c] = (__bf16)v;
  }
}

// ---------- generic bf16 WMMA GEMM: C[M,N] = A[M,K] * B[K,N] ----------
// Dynamic LDS: lsA [128x32] at byte 0, lsB [64x32] (transposed) at byte 8192.
// A tile staged with global_load_async_to_lds_b128 (ASYNCcnt path).
// epi: 0 -> bf16 raw, 1 -> f32 acc+resid, 2 -> bf16 gelu(acc+bias), 3 -> f32 acc+bias+resid
__global__ __launch_bounds__(256) void gemm_bf16_128x64(
    const __bf16* __restrict__ A, const __bf16* __restrict__ Bm,
    int M, int N, int K, void* __restrict__ out,
    const float* __restrict__ bias, const float* __restrict__ resid, int epi) {
  extern __shared__ char smem_g[];
  __bf16* lsA = (__bf16*)smem_g;            // LDS byte offset 0
  __bf16* lsB = (__bf16*)(smem_g + 8192);   // LDS byte offset 8192

  const int tid  = threadIdx.x;
  const int wave = tid >> 5, lane = tid & 31;
  const int half = (lane >> 4) & 1, l16 = lane & 15;
  const int wm = wave & 3, wn = wave >> 2;
  const int mBase = wm * 32, nBase = wn * 32;
  const long gM0 = (long)blockIdx.y * 128;
  const long gN0 = (long)blockIdx.x * 64;

  v8f acc[2][2] = {};
  const int nk = K >> 5;

  for (int kt = 0; kt < nk; ++kt) {
    // stage A tile [128 x 32]: 512 16B-chunks, 2 async copies per thread
#pragma unroll
    for (int u = 0; u < 2; ++u) {
      int ch = tid * 2 + u;
      int r = ch >> 2, cg = ch & 3;
      unsigned lds_off = (unsigned)((r * 32 + cg * 8) * 2);
      async_copy_b128(lds_off, A + (gM0 + r) * (long)K + (long)kt * 32 + cg * 8);
    }
    // stage B tile [32 x 64] transposed into [64 x 32]
    {
      int r = tid >> 3, cg = tid & 7;
      v8bf val = *(const v8bf*)(Bm + (long)(kt * 32 + r) * N + gN0 + cg * 8);
#pragma unroll
      for (int j = 0; j < 8; ++j) lsB[(cg * 8 + j) * 32 + r] = val[j];
    }
    // prefetch next A tile (global_prefetch_b8)
    if (kt + 1 < nk) {
      __builtin_prefetch(A + (gM0 + (tid >> 1)) * (long)K + (long)(kt + 1) * 32, 0, 1);
    }
    wait_asynccnt0();
    __syncthreads();

    v16bf afr[2], bfr[2];
#pragma unroll
    for (int i = 0; i < 2; ++i) afr[i] = frag_ld_bf(lsA, 32, mBase + i * 16 + l16, 0, half);
#pragma unroll
    for (int j = 0; j < 2; ++j) bfr[j] = frag_ld_bf(lsB, 32, nBase + j * 16 + l16, 0, half);
#pragma unroll
    for (int i = 0; i < 2; ++i)
#pragma unroll
      for (int j = 0; j < 2; ++j)
        acc[i][j] = __builtin_amdgcn_wmma_f32_16x16x32_bf16(
            false, afr[i], false, bfr[j], (short)0, acc[i][j], false, false);
    __syncthreads();
  }

  // epilogue; C layout: VGPR r -> M = r + 8*half within 16x16 tile, N = l16
#pragma unroll
  for (int i = 0; i < 2; ++i)
#pragma unroll
    for (int j = 0; j < 2; ++j)
#pragma unroll
      for (int r = 0; r < 8; ++r) {
        long gm = gM0 + mBase + i * 16 + half * 8 + r;
        long gn = gN0 + nBase + j * 16 + l16;
        long idx = gm * N + gn;
        float v = acc[i][j][r];
        if (epi == 0) {
          ((__bf16*)out)[idx] = (__bf16)v;
        } else if (epi == 1) {
          ((float*)out)[idx] = v + resid[idx];
        } else if (epi == 2) {
          ((__bf16*)out)[idx] = (__bf16)gelu_tanh(v + bias[gn]);
        } else {
          ((float*)out)[idx] = v + bias[gn] + resid[idx];
        }
      }
}

// ---------- block key summaries: ksum[b,n,h,e] = sum_s k[b, n*128+s, h, e] ----------
__global__ void ksum_kernel(const __bf16* __restrict__ kbf, float* __restrict__ ksum) {
  int idx = blockIdx.x * 256 + threadIdx.x; // B*NB*H*DH = 131072
  if (idx >= B_ * NB_ * H_ * DH_) return;
  int e = idx & 63, h = (idx >> 6) & 15, n = (idx >> 10) & 31, b = idx >> 15;
  const __bf16* p = kbf + ((size_t)(b * S_ + n * BLK_) * (H_ * DH_)) + h * DH_ + e;
  float s = 0.f;
  for (int si = 0; si < BLK_; ++si) s += (float)p[(size_t)si * (H_ * DH_)];
  ksum[idx] = s;
}

// ---------- logits[b,h,n,m] = sum_e ksum[b,n,h,e] * Wsort[e,m] ----------
__global__ void logits_kernel(const float* __restrict__ ksum, const float* __restrict__ Wsort,
                              float* __restrict__ logits) {
  int idx = blockIdx.x * 256 + threadIdx.x; // B*H*NB*NB = 65536
  if (idx >= B_ * H_ * NB_ * NB_) return;
  int m = idx & 31, n = (idx >> 5) & 31, h = (idx >> 10) & 15, b = idx >> 14;
  const float* ks = ksum + ((size_t)((b * NB_ + n) * H_ + h)) * DH_;
  float s = 0.f;
  for (int e = 0; e < DH_; ++e) s += ks[e] * Wsort[e * NB_ + m];
  logits[idx] = s;
}

// ---------- Sinkhorn (5 iters, log-space) in place; one wave per (b,h) ----------
__global__ __launch_bounds__(32) void sinkhorn_kernel(float* __restrict__ perm) {
  __shared__ float mat[32][33];
  const int t = threadIdx.x;
  const size_t base = (size_t)blockIdx.x * NB_ * NB_;
  for (int m = 0; m < 32; ++m) mat[t][m] = perm[base + t * 32 + m];
  __syncthreads();
  for (int it = 0; it < 5; ++it) {
    float mx = -1e30f;
    for (int m = 0; m < 32; ++m) mx = fmaxf(mx, mat[t][m]);
    float s = 0.f;
    for (int m = 0; m < 32; ++m) s += expf(mat[t][m] - mx);
    float lse = mx + logf(s);
    for (int m = 0; m < 32; ++m) mat[t][m] -= lse;
    __syncthreads();
    mx = -1e30f;
    for (int n = 0; n < 32; ++n) mx = fmaxf(mx, mat[n][t]);
    s = 0.f;
    for (int n = 0; n < 32; ++n) s += expf(mat[n][t] - mx);
    lse = mx + logf(s);
    for (int n = 0; n < 32; ++n) mat[n][t] -= lse;
    __syncthreads();
  }
  for (int m = 0; m < 32; ++m) perm[base + t * 32 + m] = expf(mat[t][m]);
}

// ---------- permute blocks: dst[b,n,s,h,e] = sum_m perm[b,h,n,m] * src[b,m*128+s,h,e] ----------
__global__ void permute_blocks(const __bf16* __restrict__ src, const float* __restrict__ perm,
                               __bf16* __restrict__ dst) {
  size_t idx = (size_t)blockIdx.x * 256 + threadIdx.x; // B*S*H*DH = 16777216
  int e = (int)(idx & 63), h = (int)((idx >> 6) & 15);
  int sg = (int)((idx >> 10) & 4095), b = (int)(idx >> 22);
  int n = sg >> 7, s = sg & 127;
  const float* pr = perm + (size_t)(((b * H_ + h) * NB_) + n) * NB_;
  float acc = 0.f;
  for (int m = 0; m < NB_; ++m) {
    acc += pr[m] * (float)src[((size_t)(b * S_ + m * BLK_ + s) * (H_ * DH_)) + h * DH_ + e];
  }
  dst[idx] = (__bf16)acc;
}

// ---------- block-local attention over [local keys ; sorted keys] ----------
// one workgroup per (b, n, h): Q[128x64] x K[256x64]^T -> softmax -> x V[256x64]
// Q and both key tiles are staged by the Tensor Data Mover (TENSORcnt path).
__global__ __launch_bounds__(256) void sortattn_kernel(
    const __bf16* __restrict__ q, const __bf16* __restrict__ k, const __bf16* __restrict__ v,
    const __bf16* __restrict__ sk, const __bf16* __restrict__ sv, __bf16* __restrict__ out) {
  extern __shared__ char smem[];
  __bf16* lsQ  = (__bf16*)smem;             // [128][64]   LDS byte 0
  __bf16* lsK  = lsQ + 128 * 64;            // [256][64]   LDS byte 16384
  __bf16* lsVt = lsK + 256 * 64;            // [64][256]   LDS byte 49152
  float*  lsS  = (float*)(lsVt + 64 * 256); // [128][256]  LDS byte 81920

  const int h = blockIdx.x & 15, n = (blockIdx.x >> 4) & 31, b = blockIdx.x >> 9;
  const size_t rowBase = (size_t)(b * S_ + n * BLK_);
  const int tid = threadIdx.x;
  const int wave = tid >> 5, lane = tid & 31;
  const int half = (lane >> 4) & 1, l16 = lane & 15;

  // wave 0 issues three TDM tile loads: Q, local keys, sorted keys
  if (wave == 0) {
    tdm_load_2d_bf16(0u,     q  + rowBase * (H_ * DH_) + h * DH_, DH_, BLK_, H_ * DH_);
    tdm_load_2d_bf16(16384u, k  + rowBase * (H_ * DH_) + h * DH_, DH_, BLK_, H_ * DH_);
    tdm_load_2d_bf16(32768u, sk + rowBase * (H_ * DH_) + h * DH_, DH_, BLK_, H_ * DH_);
  }

  // all waves stage values transposed (manual: TDM cannot transpose)
  for (int c = tid; c < 2048; c += 256) {
    int r = c >> 3, eg = c & 7;
    const __bf16* sp = (r < 128)
        ? (v  + (rowBase + r)        * (H_ * DH_) + h * DH_ + eg * 8)
        : (sv + (rowBase + (r - 128)) * (H_ * DH_) + h * DH_ + eg * 8);
    v8bf val = *(const v8bf*)sp;
#pragma unroll
    for (int j = 0; j < 8; ++j) lsVt[(eg * 8 + j) * 256 + r] = val[j];
  }
  if (wave == 0) __builtin_amdgcn_s_wait_tensorcnt(0);
  __syncthreads();

  // scores: 8x16 tiles of 16x16, 16 tiles per wave, K-depth 64 (2 steps)
  for (int tt = 0; tt < 16; ++tt) {
    int tile = wave + tt * 8;
    int mi = tile & 7, ni = tile >> 3;
    v8f acc = {};
#pragma unroll
    for (int ks = 0; ks < 2; ++ks) {
      v16bf af = frag_ld_bf(lsQ, 64, mi * 16 + l16, ks * 32, half);
      v16bf bf = frag_ld_bf(lsK, 64, ni * 16 + l16, ks * 32, half);
      acc = __builtin_amdgcn_wmma_f32_16x16x32_bf16(false, af, false, bf, (short)0, acc,
                                                    false, false);
    }
#pragma unroll
    for (int r = 0; r < 8; ++r)
      lsS[(mi * 16 + half * 8 + r) * 256 + ni * 16 + l16] = acc[r];
  }
  __syncthreads();

  // softmax rows (scale 1/sqrt(64) folded in), in place
  if (tid < 128) {
    float* row = lsS + tid * 256;
    float mx = -1e30f;
    for (int c2 = 0; c2 < 256; ++c2) mx = fmaxf(mx, row[c2] * 0.125f);
    float s = 0.f;
    for (int c2 = 0; c2 < 256; ++c2) {
      float e2 = expf(row[c2] * 0.125f - mx);
      row[c2] = e2; s += e2;
    }
    float inv = 1.0f / s;
    for (int c2 = 0; c2 < 256; ++c2) row[c2] *= inv;
  }
  __syncthreads();

  // out = P[128x256] * V[256x64]: 8x4 tiles, 4 per wave, K-depth 256 (8 steps)
  for (int tt = 0; tt < 4; ++tt) {
    int tile = wave + tt * 8;
    int mi = tile & 7, ni = tile >> 3;
    v8f acc = {};
#pragma unroll
    for (int ks = 0; ks < 8; ++ks) {
      v16bf af = frag_ld_f32(lsS, 256, mi * 16 + l16, ks * 32, half);
      v16bf bf = frag_ld_bf(lsVt, 256, ni * 16 + l16, ks * 32, half);
      acc = __builtin_amdgcn_wmma_f32_16x16x32_bf16(false, af, false, bf, (short)0, acc,
                                                    false, false);
    }
#pragma unroll
    for (int r = 0; r < 8; ++r)
      out[(rowBase + mi * 16 + half * 8 + r) * (H_ * DH_) + h * DH_ + ni * 16 + l16] =
          (__bf16)acc[r];
  }
}

// ---------- host-side orchestration ----------
extern "C" void kernel_launch(void* const* d_in, const int* in_sizes, int n_in,
                              void* d_out, int out_size, void* d_ws, size_t ws_size,
                              hipStream_t stream) {
  (void)in_sizes; (void)n_in; (void)out_size; (void)ws_size;

  const float* inputs   = (const float*)d_in[0];
  const float* Wq       = (const float*)d_in[1];
  const float* Wk       = (const float*)d_in[2];
  const float* Wv       = (const float*)d_in[3];
  const float* Wo       = (const float*)d_in[4];
  const float* Wsort    = (const float*)d_in[5];
  const float* ln1_s    = (const float*)d_in[6];
  const float* ln1_b    = (const float*)d_in[7];
  const float* ln2_s    = (const float*)d_in[8];
  const float* ln2_b    = (const float*)d_in[9];
  const float* W_mlp1   = (const float*)d_in[10];
  const float* b_mlp1   = (const float*)d_in[11];
  const float* W_mlp2   = (const float*)d_in[12];
  const float* b_mlp2   = (const float*)d_in[13];
  float* out_f32        = (float*)d_out;

  char* ws = (char*)d_ws;
  size_t off = 0;
  auto take = [&](size_t bytes) -> char* {
    char* p = ws + off;
    off += (bytes + 255) & ~(size_t)255;
    return p;
  };

  __bf16* wq_bf  = (__bf16*)take((size_t)D_ * H_ * DH_ * 2);
  __bf16* wk_bf  = (__bf16*)take((size_t)D_ * H_ * DH_ * 2);
  __bf16* wv_bf  = (__bf16*)take((size_t)D_ * H_ * DH_ * 2);
  __bf16* wo_bf  = (__bf16*)take((size_t)H_ * DH_ * D_ * 2);
  __bf16* w1_bf  = (__bf16*)take((size_t)D_ * 4 * D_ * 2);
  __bf16* w2_bf  = (__bf16*)take((size_t)4 * D_ * D_ * 2);
  __bf16* xln_bf = (__bf16*)take((size_t)MTOK * D_ * 2);
  __bf16* q_bf   = (__bf16*)take((size_t)MTOK * D_ * 2);
  __bf16* k_bf   = (__bf16*)take((size_t)MTOK * D_ * 2);
  __bf16* v_bf   = (__bf16*)take((size_t)MTOK * D_ * 2);
  __bf16* sk_bf  = (__bf16*)take((size_t)MTOK * D_ * 2);
  __bf16* sv_bf  = (__bf16*)take((size_t)MTOK * D_ * 2);
  __bf16* ao_bf  = (__bf16*)take((size_t)MTOK * D_ * 2);
  __bf16* yln_bf = (__bf16*)take((size_t)MTOK * D_ * 2);
  float*  ksum   = (float*)take((size_t)B_ * NB_ * H_ * DH_ * 4);
  float*  perm   = (float*)take((size_t)B_ * H_ * NB_ * NB_ * 4);
  float*  xres   = (float*)take((size_t)MTOK * D_ * 4);
  __bf16* h1_bf  = (__bf16*)take((size_t)MTOK * 4 * D_ * 2);

  const size_t gemm_smem = (size_t)(128 * 32 + 64 * 32) * sizeof(__bf16); // 12 KB

  // --- weights to bf16 ---
  cvt_f32_bf16<<<4096, 256, 0, stream>>>(Wq, wq_bf, D_ * H_ * DH_);
  cvt_f32_bf16<<<4096, 256, 0, stream>>>(Wk, wk_bf, D_ * H_ * DH_);
  cvt_f32_bf16<<<4096, 256, 0, stream>>>(Wv, wv_bf, D_ * H_ * DH_);
  cvt_f32_bf16<<<4096, 256, 0, stream>>>(Wo, wo_bf, H_ * DH_ * D_);
  cvt_f32_bf16<<<16384, 256, 0, stream>>>(W_mlp1, w1_bf, D_ * 4 * D_);
  cvt_f32_bf16<<<16384, 256, 0, stream>>>(W_mlp2, w2_bf, 4 * D_ * D_);

  // --- pre-LN ---
  layernorm_bf16<<<MTOK, 256, 0, stream>>>(inputs, ln1_s, ln1_b, xln_bf);

  // --- QKV projections (WMMA bf16, async-LDS staging) ---
  dim3 gQKV(D_ / 64, MTOK / 128);
  gemm_bf16_128x64<<<gQKV, 256, gemm_smem, stream>>>(xln_bf, wq_bf, MTOK, D_, D_, q_bf, nullptr, nullptr, 0);
  gemm_bf16_128x64<<<gQKV, 256, gemm_smem, stream>>>(xln_bf, wk_bf, MTOK, D_, D_, k_bf, nullptr, nullptr, 0);
  gemm_bf16_128x64<<<gQKV, 256, gemm_smem, stream>>>(xln_bf, wv_bf, MTOK, D_, D_, v_bf, nullptr, nullptr, 0);

  // --- sorting network ---
  ksum_kernel<<<(B_ * NB_ * H_ * DH_) / 256, 256, 0, stream>>>(k_bf, ksum);
  logits_kernel<<<(B_ * H_ * NB_ * NB_) / 256, 256, 0, stream>>>(ksum, Wsort, perm);
  sinkhorn_kernel<<<B_ * H_, 32, 0, stream>>>(perm);

  // --- permute key/value blocks ---
  permute_blocks<<<(MTOK * D_) / 256, 256, 0, stream>>>(k_bf, perm, sk_bf);
  permute_blocks<<<(MTOK * D_) / 256, 256, 0, stream>>>(v_bf, perm, sv_bf);

  // --- block-local attention (TDM staging + WMMA) ---
  size_t smem_bytes = (size_t)(128 * 64 + 256 * 64 + 64 * 256) * sizeof(__bf16) +
                      (size_t)128 * 256 * sizeof(float); // 208 KB
  sortattn_kernel<<<B_ * NB_ * H_, 256, smem_bytes, stream>>>(q_bf, k_bf, v_bf, sk_bf, sv_bf, ao_bf);

  // --- output projection + residual ---
  gemm_bf16_128x64<<<dim3(D_ / 64, MTOK / 128), 256, gemm_smem, stream>>>(
      ao_bf, wo_bf, MTOK, D_, D_, xres, nullptr, inputs, 1);

  // --- MLP block ---
  layernorm_bf16<<<MTOK, 256, 0, stream>>>(xres, ln2_s, ln2_b, yln_bf);
  gemm_bf16_128x64<<<dim3(4 * D_ / 64, MTOK / 128), 256, gemm_smem, stream>>>(
      yln_bf, w1_bf, MTOK, 4 * D_, D_, h1_bf, b_mlp1, nullptr, 2);
  gemm_bf16_128x64<<<dim3(D_ / 64, MTOK / 128), 256, gemm_smem, stream>>>(
      h1_bf, w2_bf, MTOK, D_, 4 * D_, out_f32, b_mlp2, xres, 3);
}
